// Phi3AttentionExperimental_65893388255940
// MI455X (gfx1250) — matrix operand order repaired and checked by reading, not compiled, not run on hardware
//
#include <hip/hip_runtime.h>
#include <cstdint>
#include <cstddef>

// ---------------------------------------------------------------------------
// Phi3 sparse attention for MI455X (gfx1250, wave32, WMMA + TDM).
// B=2 S=1024 HID=3072 H=32 D=96 DD=16, sparse=0.5, min_sparse=4
// ---------------------------------------------------------------------------

typedef __attribute__((ext_vector_type(16))) __bf16    v16bf;
typedef __attribute__((ext_vector_type(8)))  float     v8f;
typedef __attribute__((ext_vector_type(2)))  float     v2f;
typedef __attribute__((ext_vector_type(4)))  uint32_t  u32x4;
typedef __attribute__((ext_vector_type(8)))  uint32_t  u32x8;

#define B_    2
#define S_    1024
#define HID_  3072
#define H_    32
#define D_    96
#define DD_   16
#define NEG_  (-1.0e30f)

// ---------------------------------------------------------------------------
// Tensor Data Mover: DMA a 16-row x 32-col bf16 strip (global, row stride =
// `stride0` elements) into LDS at byte offset `lds_off`, inserting 1 DWORD of
// LDS padding after every 16 DWORDs (64B row) so the LDS row stride becomes
// 34 bf16 = 68 B (17-bank stride, conflict-free WMMA fragment reads).
// Issued once per wave (TDM ignores EXEC); tracked by TENSORcnt.
// ---------------------------------------------------------------------------
__device__ __forceinline__ void tdm_load_strip_16x32(
    uint32_t lds_off, const __bf16* gsrc, uint32_t stride0_elems) {
  uint64_t ga = (uint64_t)(uintptr_t)gsrc;
  // D# group 0: count=1 | lds_addr | global_addr[56:0] | type=2
  u32x4 g0 = {
      1u,                                                   // count=1, user D#
      lds_off,                                              // LDS byte address
      (uint32_t)ga,                                         // global_addr[31:0]
      ((uint32_t)(ga >> 32) & 0x01FFFFFFu) | (2u << 30)     // ga[56:32], type=2
  };
  // D# group 1: data_size=2B, pad_enable, pad_interval=16 DW, pad_amount=1 DW,
  // tensor_dim0=32, tensor_dim1=16, tile_dim0=32, tile_dim1=16, stride0.
  const uint32_t tdim0 = 32u, tdim1 = 16u;
  u32x8 g1 = {
      (1u << 16) | (1u << 20) | (3u << 22),                 // dsize|pad|interval
      (tdim0 & 0xFFFFu) << 16,                              // tensor_dim0 lo16
      (tdim0 >> 16) | ((tdim1 & 0xFFFFu) << 16),            // dim0 hi | dim1 lo
      (tdim1 >> 16) | (32u << 16),                          // dim1 hi | tile_dim0
      16u,                                                  // tile_dim1 (tile_dim2=0)
      stride0_elems,                                        // tensor_dim0_stride lo32
      0u,                                                   // stride0 hi | stride1 lo
      0u
  };
  asm volatile("tensor_load_to_lds %0, %1" :: "s"(g0), "s"(g1) : "memory");
}

// ---------------------------------------------------------------------------
// f32 -> bf16 conversion
// ---------------------------------------------------------------------------
__global__ void cvt_f32_bf16(const float* __restrict__ in,
                             __bf16* __restrict__ out, long n) {
  long i = (long)blockIdx.x * blockDim.x + threadIdx.x;
  if (i < n) out[i] = (__bf16)in[i];
}

// ---------------------------------------------------------------------------
// bf16 GEMM: C[M,N](f32) = A[M,K](bf16) @ B[N,K](bf16)^T
// Workgroup = 128 threads (4 waves), tile 64x64, K-step 32.
// Tile staging via Tensor Data Mover (one A strip + one B strip per wave),
// compute via v_wmma_f32_16x16x32_bf16.
// ---------------------------------------------------------------------------
__global__ __launch_bounds__(128) void gemm_bf16_nt(
    const __bf16* __restrict__ A, const __bf16* __restrict__ Bm,
    float* __restrict__ C, int M, int N, int K) {
  __shared__ __bf16 As[64][34];   // 34 = 32 + TDM pad DWORD (17-bank stride)
  __shared__ __bf16 Bs[64][34];

  const int t    = threadIdx.x;
  const int lane = t & 31;
  const int w    = t >> 5;
  const int half = lane >> 4;
  const int ln   = lane & 15;
  const int m0   = blockIdx.y * 64;
  const int n0   = blockIdx.x * 64;

  const uint32_t as_base = (uint32_t)(uintptr_t)&As[0][0];
  const uint32_t bs_base = (uint32_t)(uintptr_t)&Bs[0][0];
  const uint32_t strip   = (uint32_t)(16 * 34 * 2);   // 1088 B per wave strip

  v8f acc[4] = {};

  for (int k0 = 0; k0 < K; k0 += 32) {
    __syncthreads();   // previous iteration's fragment reads complete
    // Each wave DMAs its 16-row strip of the A and B tiles into LDS.
    tdm_load_strip_16x32(as_base + (uint32_t)w * strip,
                         A + (size_t)(m0 + w * 16) * K + k0, (uint32_t)K);
    tdm_load_strip_16x32(bs_base + (uint32_t)w * strip,
                         Bm + (size_t)(n0 + w * 16) * K + k0, (uint32_t)K);
    __builtin_amdgcn_s_wait_tensorcnt(0);
    __syncthreads();   // publish all strips to all waves

    // A fragment (16x32): VGPR v holds K = 16*(v/4) + 8*half + 2*(v%4) (+1)
    v16bf af;
#pragma unroll
    for (int v = 0; v < 8; ++v) {
      int kk = (v >> 2) * 16 + half * 8 + (v & 3) * 2;
      af[2 * v]     = As[w * 16 + ln][kk];
      af[2 * v + 1] = As[w * 16 + ln][kk + 1];
    }
#pragma unroll
    for (int nt = 0; nt < 4; ++nt) {
      // B fragment (32x16): VGPR v holds K = 16*(v/4) + 8*half + 2*(v%4) (+1), N = ln
      v16bf bf;
#pragma unroll
      for (int v = 0; v < 8; ++v) {
        int kk = (v >> 2) * 16 + half * 8 + (v & 3) * 2;
        bf[2 * v]     = Bs[nt * 16 + ln][kk];
        bf[2 * v + 1] = Bs[nt * 16 + ln][kk + 1];
      }
      acc[nt] = __builtin_amdgcn_wmma_f32_16x16x32_bf16(
          false, af, false, bf, (short)0, acc[nt], false, false);
    }
  }

  // C/D layout: VGPR r -> M = r + 8*half, N = ln
#pragma unroll
  for (int nt = 0; nt < 4; ++nt) {
#pragma unroll
    for (int r = 0; r < 8; ++r) {
      int mrow = m0 + w * 16 + r + 8 * half;
      int ncol = n0 + nt * 16 + ln;
      C[(size_t)mrow * N + ncol] = acc[nt][r];
    }
  }
}

// ---------------------------------------------------------------------------
// RoPE + head split: qkv f32 [B,S,3*HID] -> q,k (roped), v  bf16 [B,H,S,D]
// ---------------------------------------------------------------------------
__global__ void rope_split_kernel(const float* __restrict__ qkv,
                                  const int* __restrict__ pos,
                                  __bf16* __restrict__ q,
                                  __bf16* __restrict__ k,
                                  __bf16* __restrict__ v) {
  size_t i = (size_t)blockIdx.x * blockDim.x + threadIdx.x;
  const size_t total = (size_t)B_ * H_ * S_ * D_;
  if (i >= total) return;
  int d = (int)(i % D_);
  int s = (int)((i / D_) % S_);
  int h = (int)((i / ((size_t)D_ * S_)) % H_);
  int b = (int)(i / ((size_t)D_ * S_ * H_));

  float p   = (float)pos[b * S_ + s];
  int   fi  = d % 48;                                 // inv_freq index
  float invf = __expf(-(2.0f * (float)fi / (float)D_) * __logf(10000.0f));
  float ang = p * invf;
  float cs = __cosf(ang), sn = __sinf(ang);

  size_t row = (size_t)(b * S_ + s) * (3 * HID_);
  int col  = h * D_ + d;
  int pcol = (d < 48) ? col + 48 : col - 48;          // rotate_half partner
  float sgn = (d < 48) ? -1.0f : 1.0f;

  float qx = qkv[row + col];
  float kx = qkv[row + HID_ + col];
  float vx = qkv[row + 2 * HID_ + col];
  float qp = qkv[row + pcol];
  float kp = qkv[row + HID_ + pcol];

  q[i] = (__bf16)(qx * cs + sgn * qp * sn);
  k[i] = (__bf16)(kx * cs + sgn * kp * sn);
  v[i] = (__bf16)vx;
}

// ---------------------------------------------------------------------------
// Importance logits (v_wmma_f32_16x16x4_f32, K=16 via 4 chained WMMAs) +
// exact stable top-k rank -> keep bitmask (128 bytes per (g,q) row).
// Softmax skipped: it is monotonic, ranks are identical on raw logits.
// Workgroup: 128 threads handles 16 queries x all causal keys.
// ---------------------------------------------------------------------------
__global__ __launch_bounds__(128) void imp_mask_kernel(
    const float* __restrict__ q_imp, const float* __restrict__ k_imp,
    unsigned char* __restrict__ keep) {
  __shared__ float simp[16][S_];    // 64KB importance logits for 16 queries

  const int t    = threadIdx.x;
  const int lane = t & 31;
  const int w    = t >> 5;
  const int half = lane >> 4;
  const int ln   = lane & 15;
  const int qt   = blockIdx.x;      // query tile (0..63)
  const int g    = blockIdx.y;      // b*H + h    (0..63)
  const int q0   = qt * 16;

  // A fragments: f32 16x4 layout -> VGPR v holds K = koff + 2*half + v, M = ln
  v2f af[4];
#pragma unroll
  for (int ko = 0; ko < 4; ++ko) {
    const float* qp = q_imp + ((size_t)g * S_ + q0 + ln) * DD_ + ko * 4 + 2 * half;
    af[ko] = v2f{qp[0], qp[1]};
  }

  for (int kt = w; kt <= qt; kt += 4) {   // causal key tiles, striped by wave
    int k0 = kt * 16;
    v8f c = {};
#pragma unroll
    for (int ko = 0; ko < 4; ++ko) {
      const float* kp = k_imp + ((size_t)g * S_ + k0 + ln) * DD_ + ko * 4 + 2 * half;
      v2f bf = v2f{kp[0], kp[1]};
      c = __builtin_amdgcn_wmma_f32_16x16x4_f32(
          false, af[ko], false, bf, (short)0, c, false, false);
    }
#pragma unroll
    for (int r = 0; r < 8; ++r)
      simp[r + 8 * half][k0 + ln] = c[r];
  }
  __syncthreads();

  // Exact stable rank: rank(k) = #{j<=q : v_j > v_k} + #{j<k : v_j == v_k}
  for (int m = 0; m < 16; ++m) {
    int q = q0 + m;
    int keep_cnt = (q + 2) >> 1;            // ceil(0.5*(q+1))
    if (keep_cnt < 4) keep_cnt = 4;
    unsigned byte = 0;
#pragma unroll
    for (int kk = 0; kk < 8; ++kk) {
      int k = t * 8 + kk;
      if (k <= q) {
        bool kb;
        if (k < 4) {
          kb = true;                         // sink tokens always kept
        } else {
          float vk = simp[m][k];
          int cnt = 0;
          for (int j = 0; j <= q; ++j) {     // broadcast LDS read
            float vj = simp[m][j];
            cnt += ((vj > vk) || (vj == vk && j < k)) ? 1 : 0;
            if (cnt >= keep_cnt) break;
          }
          kb = cnt < keep_cnt;
        }
        if (kb) byte |= 1u << kk;
      }
    }
    keep[((size_t)g * S_ + q) * 128 + t] = (unsigned char)byte;
  }
}

// ---------------------------------------------------------------------------
// Sparse causal flash attention. One wave per 16-query tile; 4 waves / WG.
// QK^T: 3x v_wmma_f32_16x16x32_bf16 (D=96). PV: 6x bf16 WMMA (K=32 keys).
// Online softmax with wave32 shfl_xor row reductions (within 16-lane group).
// global_prefetch covers the next key/value block.
// ---------------------------------------------------------------------------
__global__ __launch_bounds__(128) void attn_kernel(
    const __bf16* __restrict__ Q, const __bf16* __restrict__ Kt,
    const __bf16* __restrict__ V, const unsigned char* __restrict__ keep,
    __bf16* __restrict__ ctx) {
  __shared__ __bf16 p_lds[4][16][34];       // per-wave P transpose buffer

  const int t    = threadIdx.x;
  const int lane = t & 31;
  const int w    = t >> 5;
  const int half = lane >> 4;
  const int ln   = lane & 15;
  const int b    = blockIdx.z;
  const int h    = blockIdx.y;
  const int qt   = blockIdx.x * 4 + w;
  const int q0   = qt * 16;
  const int g    = b * H_ + h;

  const __bf16* Qp = Q + (size_t)g * S_ * D_;
  const __bf16* Kp = Kt + (size_t)g * S_ * D_;
  const __bf16* Vp = V + (size_t)g * S_ * D_;

  // Loop-invariant Q fragments, 3 K=32 chunks over D=96
  v16bf aq[3];
#pragma unroll
  for (int c = 0; c < 3; ++c) {
#pragma unroll
    for (int v = 0; v < 8; ++v) {
      int kk = (v >> 2) * 16 + half * 8 + (v & 3) * 2;
      aq[c][2 * v]     = Qp[(size_t)(q0 + ln) * D_ + c * 32 + kk];
      aq[c][2 * v + 1] = Qp[(size_t)(q0 + ln) * D_ + c * 32 + kk + 1];
    }
  }

  v8f oacc[6] = {};
  float row_m[8], row_l[8];
#pragma unroll
  for (int r = 0; r < 8; ++r) { row_m[r] = -3.0e38f; row_l[r] = 0.0f; }

  const float scale = 0.1020620726f;        // 1/sqrt(96)
  const int nblk = (qt + 2) >> 1;           // 32-key blocks covering [0, q0+15]

  for (int blk = 0; blk < nblk; ++blk) {
    int k0 = blk * 32;
    if (blk + 1 < nblk) {                   // prefetch next K/V block rows
      __builtin_prefetch(Kp + (size_t)(k0 + 32 + lane) * D_, 0, 0);
      __builtin_prefetch(Vp + (size_t)(k0 + 32 + lane) * D_, 0, 0);
    }
    float p0[8], p1[8];

#pragma unroll
    for (int sub = 0; sub < 2; ++sub) {
      int ks = k0 + sub * 16;
      v8f cs = {};
#pragma unroll
      for (int c = 0; c < 3; ++c) {
        v16bf bk;
#pragma unroll
        for (int v = 0; v < 8; ++v) {
          int kk = (v >> 2) * 16 + half * 8 + (v & 3) * 2;
          bk[2 * v]     = Kp[(size_t)(ks + ln) * D_ + c * 32 + kk];
          bk[2 * v + 1] = Kp[(size_t)(ks + ln) * D_ + c * 32 + kk + 1];
        }
        cs = __builtin_amdgcn_wmma_f32_16x16x32_bf16(
            false, aq[c], false, bk, (short)0, cs, false, false);
      }
      int key   = ks + ln;                  // key index (fixed per lane)
      int kbyte = key >> 3, kbit = key & 7;
      float* pp = sub ? p1 : p0;
#pragma unroll
      for (int r = 0; r < 8; ++r) {
        int qrow = q0 + r + 8 * half;
        float s = cs[r] * scale;
        bool ok = (key <= qrow);            // causal
        if (ok && key >= 4) {               // learned sparsity mask
          unsigned bv = keep[((size_t)g * S_ + qrow) * 128 + kbyte];
          ok = (bv >> kbit) & 1u;
        }
        pp[r] = ok ? s : NEG_;
      }
    }

    // Online softmax update (rows live across the 16 lanes of each half)
#pragma unroll
    for (int r = 0; r < 8; ++r) {
      float mx = fmaxf(p0[r], p1[r]);
#pragma unroll
      for (int off = 1; off < 16; off <<= 1)
        mx = fmaxf(mx, __shfl_xor(mx, off, 32));
      float mnew = fmaxf(row_m[r], mx);
      float scl  = __expf(row_m[r] - mnew);
      float e0   = __expf(p0[r] - mnew);
      float e1   = __expf(p1[r] - mnew);
      float ps   = e0 + e1;
#pragma unroll
      for (int off = 1; off < 16; off <<= 1)
        ps += __shfl_xor(ps, off, 32);
      row_l[r] = row_l[r] * scl + ps;
      row_m[r] = mnew;
#pragma unroll
      for (int dt = 0; dt < 6; ++dt) oacc[dt][r] *= scl;
      p_lds[w][r + 8 * half][ln]      = (__bf16)e0;
      p_lds[w][r + 8 * half][ln + 16] = (__bf16)e1;
    }

    // P fragment (A 16x32) read back transposed from LDS (same wave -> in order)
    v16bf ap;
#pragma unroll
    for (int v = 0; v < 8; ++v) {
      int kk = (v >> 2) * 16 + half * 8 + (v & 3) * 2;
      ap[2 * v]     = p_lds[w][ln][kk];
      ap[2 * v + 1] = p_lds[w][ln][kk + 1];
    }
#pragma unroll
    for (int dt = 0; dt < 6; ++dt) {
      v16bf bv;
#pragma unroll
      for (int v = 0; v < 8; ++v) {
        int kk = (v >> 2) * 16 + half * 8 + (v & 3) * 2;       // key row within 32-block
        bv[2 * v]     = Vp[(size_t)(k0 + kk) * D_ + dt * 16 + ln];
        bv[2 * v + 1] = Vp[(size_t)(k0 + kk + 1) * D_ + dt * 16 + ln];
      }
      oacc[dt] = __builtin_amdgcn_wmma_f32_16x16x32_bf16(
          false, ap, false, bv, (short)0, oacc[dt], false, false);
    }
  }

  // Epilogue: ctx [B,S,H*D] bf16 (A matrix of the O-projection GEMM)
#pragma unroll
  for (int r = 0; r < 8; ++r) {
    int qrow = q0 + r + 8 * half;
    float inv = 1.0f / row_l[r];
#pragma unroll
    for (int dt = 0; dt < 6; ++dt)
      ctx[((size_t)(b * S_ + qrow)) * HID_ + h * D_ + dt * 16 + ln] =
          (__bf16)(oacc[dt][r] * inv);
  }
}

// ---------------------------------------------------------------------------
// Launch
// ---------------------------------------------------------------------------
extern "C" void kernel_launch(void* const* d_in, const int* in_sizes, int n_in,
                              void* d_out, int out_size, void* d_ws,
                              size_t ws_size, hipStream_t stream) {
  (void)in_sizes; (void)n_in; (void)out_size; (void)ws_size;

  const float* hidden = (const float*)d_in[0];   // [B,S,HID]
  const float* qkv_w  = (const float*)d_in[1];   // [3*HID, HID]
  const float* o_w    = (const float*)d_in[2];   // [HID, HID]
  const float* q_imp  = (const float*)d_in[3];   // [B*H, S, DD]
  const float* k_imp  = (const float*)d_in[4];   // [B*H, S, DD]
  const int*   pos    = (const int*)d_in[5];     // [B,S]

  char* base = (char*)d_ws;
  size_t off = 0;
  auto carve = [&](size_t bytes) -> char* {
    char* p = base + off;
    off = (off + bytes + 255) & ~(size_t)255;
    return p;
  };

  __bf16* hid_bf  = (__bf16*)carve((size_t)B_ * S_ * HID_ * 2);
  __bf16* qkvw_bf = (__bf16*)carve((size_t)3 * HID_ * HID_ * 2);
  __bf16* ow_bf   = (__bf16*)carve((size_t)HID_ * HID_ * 2);
  float*  qkv_f   = (float*)carve((size_t)B_ * S_ * 3 * HID_ * 4);
  __bf16* qb      = (__bf16*)carve((size_t)B_ * H_ * S_ * D_ * 2);
  __bf16* kb      = (__bf16*)carve((size_t)B_ * H_ * S_ * D_ * 2);
  __bf16* vb      = (__bf16*)carve((size_t)B_ * H_ * S_ * D_ * 2);
  unsigned char* keepm = (unsigned char*)carve((size_t)B_ * H_ * S_ * 128);
  __bf16* ctx     = (__bf16*)carve((size_t)B_ * S_ * HID_ * 2);

  const long n_hid = (long)B_ * S_ * HID_;
  const long n_qkw = (long)3 * HID_ * HID_;
  const long n_ow  = (long)HID_ * HID_;

  cvt_f32_bf16<<<(n_hid + 255) / 256, 256, 0, stream>>>(hidden, hid_bf, n_hid);
  cvt_f32_bf16<<<(n_qkw + 255) / 256, 256, 0, stream>>>(qkv_w, qkvw_bf, n_qkw);
  cvt_f32_bf16<<<(n_ow + 255) / 256, 256, 0, stream>>>(o_w, ow_bf, n_ow);

  // QKV projection: [2048,9216] = hid_bf [2048,3072] @ qkv_w^T
  gemm_bf16_nt<<<dim3((3 * HID_) / 64, (B_ * S_) / 64), 128, 0, stream>>>(
      hid_bf, qkvw_bf, qkv_f, B_ * S_, 3 * HID_, HID_);

  rope_split_kernel<<<((long)B_ * H_ * S_ * D_ + 255) / 256, 256, 0, stream>>>(
      qkv_f, pos, qb, kb, vb);

  imp_mask_kernel<<<dim3(S_ / 16, B_ * H_), 128, 0, stream>>>(q_imp, k_imp, keepm);

  attn_kernel<<<dim3(S_ / 64, H_, B_), 128, 0, stream>>>(qb, kb, vb, keepm, ctx);

  // Output projection into d_out (f32)
  gemm_bf16_nt<<<dim3(HID_ / 64, (B_ * S_) / 64), 128, 0, stream>>>(
      ctx, ow_bf, (float*)d_out, B_ * S_, HID_, HID_);
}
